// NCB_76965813944530
// MI455X (gfx1250) — compile-verified, matrix operand order checked by reading, not run on hardware
//
#include <hip/hip_runtime.h>
#include <hip/hip_bf16.h>

typedef __attribute__((ext_vector_type(16))) __bf16   v16bf;
typedef __attribute__((ext_vector_type(8)))  float    v8f;
typedef __attribute__((ext_vector_type(8)))  unsigned int v8u;

__device__ __forceinline__ unsigned short f2bf(float f) {
  unsigned int u = __builtin_bit_cast(unsigned int, f);
  u += 0x7FFFu + ((u >> 16) & 1u);          // round-to-nearest-even
  return (unsigned short)(u >> 16);
}

// ---------------------------------------------------------------------------
// C[M,N] = act(A[M,K] @ B[K,N] + bias), A/B f32 row-major in HBM, converted to
// bf16 while staging into LDS; f32 accumulate via v_wmma_f32_16x16x32_bf16.
// Block: 256 threads = 8 waves; tile 128(M) x 128(N) x 32(K).
// Waves: 4 along M x 2 along N; each wave owns a 32x64 region = 2x4 WMMA
// subtiles (8 v_wmma per k-step vs 6 fragment loads).
// ---------------------------------------------------------------------------
#define MT 128
#define NT 128
#define KT 32
#define LDA 34   // bf16 elems per LDS row (17 dwords, odd -> conflict-free)
#define LDB 34

__global__ __launch_bounds__(256) void gemm_bf16_kernel(
    const float* __restrict__ A, const float* __restrict__ B,
    const float* __restrict__ bias, float* __restrict__ C,
    int M, int N, int K, int relu)
{
  __shared__ unsigned short As[MT * LDA];   // 128 x 32 (K-major)
  __shared__ unsigned short Bs[NT * LDB];   // 128 x 32 (transposed: [n][k])

  const int tid  = threadIdx.x;
  const int lane = tid & 31;
  const int wave = tid >> 5;
  const int wm   = wave & 3;     // 4 waves along M (32 rows each)
  const int wn   = wave >> 2;    // 2 waves along N (64 cols each)
  const int bm   = blockIdx.x * MT;
  const int bn   = blockIdx.y * NT;

  v8f acc[2][4] = {};

  const int l16 = lane & 15;
  const int dk  = (lane >> 4) * 4;   // dword offset of this lane's K-half

  const int ar = tid >> 3;           // A staging: rows ar + {0,32,64,96}
  const int ac = (tid & 7) * 4;      //            cols 0..28 step 4 (float4)
  const int br = tid >> 5;           // B staging: k = br + {0,8,16,24}
  const int bc = (tid & 31) * 4;     //            n 0..124 step 4 (float4)

  for (int k0 = 0; k0 < K; k0 += KT) {
    // ---- stage tiles: batch all global loads, then convert+store ----
    float4 va[4], vb[4];
    #pragma unroll
    for (int it = 0; it < 4; ++it)
      va[it] = *(const float4*)(A + (size_t)(bm + ar + it * 32) * K + (k0 + ac));
    #pragma unroll
    for (int it = 0; it < 4; ++it)
      vb[it] = *(const float4*)(B + (size_t)(k0 + br + it * 8) * N + (bn + bc));
    if (k0 + KT < K) {
      __builtin_prefetch(A + (size_t)(bm + ar) * K + (k0 + KT + ac), 0, 1);
      __builtin_prefetch(B + (size_t)(k0 + KT + br) * N + (bn + bc), 0, 1);
    }
    #pragma unroll
    for (int it = 0; it < 4; ++it) {
      unsigned short* d = &As[(ar + it * 32) * LDA + ac];
      d[0] = f2bf(va[it].x); d[1] = f2bf(va[it].y);
      d[2] = f2bf(va[it].z); d[3] = f2bf(va[it].w);
    }
    #pragma unroll
    for (int it = 0; it < 4; ++it) {
      const int kk = br + it * 8;
      Bs[(bc + 0) * LDB + kk] = f2bf(vb[it].x);
      Bs[(bc + 1) * LDB + kk] = f2bf(vb[it].y);
      Bs[(bc + 2) * LDB + kk] = f2bf(vb[it].z);
      Bs[(bc + 3) * LDB + kk] = f2bf(vb[it].w);
    }
    __syncthreads();

    // ---- fragments per ISA 16-bit 16x32 layout ----
    v16bf afrag[2], bfrag[4];
    #pragma unroll
    for (int i = 0; i < 2; ++i) {
      const unsigned int* p =
          (const unsigned int*)&As[(wm * 32 + i * 16 + l16) * LDA];
      v8u t;
      t[0] = p[dk + 0]; t[1] = p[dk + 1]; t[2] = p[dk + 2]; t[3] = p[dk + 3];
      t[4] = p[8 + dk + 0]; t[5] = p[8 + dk + 1];
      t[6] = p[8 + dk + 2]; t[7] = p[8 + dk + 3];
      afrag[i] = __builtin_bit_cast(v16bf, t);
    }
    #pragma unroll
    for (int j = 0; j < 4; ++j) {
      const unsigned int* p =
          (const unsigned int*)&Bs[(wn * 64 + j * 16 + l16) * LDB];
      v8u t;
      t[0] = p[dk + 0]; t[1] = p[dk + 1]; t[2] = p[dk + 2]; t[3] = p[dk + 3];
      t[4] = p[8 + dk + 0]; t[5] = p[8 + dk + 1];
      t[6] = p[8 + dk + 2]; t[7] = p[8 + dk + 3];
      bfrag[j] = __builtin_bit_cast(v16bf, t);
    }
    #pragma unroll
    for (int i = 0; i < 2; ++i)
      #pragma unroll
      for (int j = 0; j < 4; ++j)
        acc[i][j] = __builtin_amdgcn_wmma_f32_16x16x32_bf16(
            false, afrag[i], false, bfrag[j], (short)0, acc[i][j], false, false);
    __syncthreads();
  }

  // Epilogue: acc element v is (M = base + v + 8*(lane>=16), N = base + l16)
  const int hi = (lane >> 4) * 8;
  #pragma unroll
  for (int j = 0; j < 4; ++j) {
    const int col = bn + wn * 64 + j * 16 + l16;
    const float bv = bias ? bias[col] : 0.0f;
    #pragma unroll
    for (int i = 0; i < 2; ++i) {
      const int r0 = bm + wm * 32 + i * 16 + hi;
      #pragma unroll
      for (int v = 0; v < 8; ++v) {
        float x = acc[i][j][v] + bv;
        if (relu) x = fmaxf(x, 0.0f);
        C[(size_t)(r0 + v) * N + col] = x;
      }
    }
  }
}

// ---------------------------------------------------------------------------
// Row LayerNorm: out = LN(relu?(in)) * g + b (+ add). One block (256t) per row.
// ---------------------------------------------------------------------------
__global__ __launch_bounds__(256) void ln_kernel(
    const float* __restrict__ in, const float* __restrict__ add,
    float* __restrict__ out, const float* __restrict__ g,
    const float* __restrict__ b, int H, int relu_in)
{
  __shared__ float red[256];
  const int row = blockIdx.x;
  const int tid = threadIdx.x;
  const float* x = in + (size_t)row * H;
  float vals[4];
  int n = 0;
  float s = 0.f;
  for (int c = tid; c < H; c += 256) {
    float v = x[c];
    if (relu_in) v = fmaxf(v, 0.f);
    vals[n++] = v;
    s += v;
  }
  red[tid] = s; __syncthreads();
  for (int st = 128; st > 0; st >>= 1) {
    if (tid < st) red[tid] += red[tid + st];
    __syncthreads();
  }
  const float mean = red[0] / (float)H;
  __syncthreads();
  float sq = 0.f;
  for (int k = 0; k < n; ++k) { float d = vals[k] - mean; sq += d * d; }
  red[tid] = sq; __syncthreads();
  for (int st = 128; st > 0; st >>= 1) {
    if (tid < st) red[tid] += red[tid + st];
    __syncthreads();
  }
  const float rstd = rsqrtf(red[0] / (float)H + 1e-5f);
  n = 0;
  for (int c = tid; c < H; c += 256) {
    float o = (vals[n++] - mean) * rstd * g[c] + b[c];
    if (add) o += add[(size_t)row * H + c];
    out[(size_t)row * H + c] = o;
  }
}

// aw[i] = dot(sigmoid(g1[i,:]) * tanh(g2[i,:]), a3w) + a3b; s[i] = sigmoid(aw)
__global__ __launch_bounds__(256) void aw_kernel(
    const float* __restrict__ g1, const float* __restrict__ g2,
    const float* __restrict__ a3w, const float* __restrict__ a3b,
    float* __restrict__ aw_out, float* __restrict__ s_out, int H)
{
  __shared__ float red[256];
  const int row = blockIdx.x;
  const int tid = threadIdx.x;
  float acc = 0.f;
  for (int c = tid; c < H; c += 256) {
    const size_t idx = (size_t)row * H + c;
    const float u = 1.f / (1.f + expf(-g1[idx]));
    const float t = tanhf(g2[idx]);
    acc += u * t * a3w[c];
  }
  red[tid] = acc; __syncthreads();
  for (int st = 128; st > 0; st >>= 1) {
    if (tid < st) red[tid] += red[tid + st];
    __syncthreads();
  }
  if (tid == 0) {
    const float a = red[0] + a3b[0];
    aw_out[row] = a;
    s_out[row] = 1.f / (1.f + expf(-a));
  }
}

// Deterministic single-block sum of n floats.
__global__ __launch_bounds__(256) void sum_kernel(
    const float* __restrict__ v, float* __restrict__ out, int n)
{
  __shared__ float red[256];
  float a = 0.f;
  for (int i = threadIdx.x; i < n; i += 256) a += v[i];
  red[threadIdx.x] = a; __syncthreads();
  for (int st = 128; st > 0; st >>= 1) {
    if (threadIdx.x < st) red[threadIdx.x] += red[threadIdx.x + st];
    __syncthreads();
  }
  if (threadIdx.x == 0) out[0] = red[0];
}

// dinv[i] = rsqrt(s_i*(S - s_i) + 1);  t[i] = s_i * dinv[i]
__global__ void scal_kernel(const float* __restrict__ s,
                            const float* __restrict__ Ssum,
                            float* __restrict__ dinv, float* __restrict__ t,
                            int N)
{
  const int i = blockIdx.x * blockDim.x + threadIdx.x;
  if (i < N) {
    const float sv = s[i];
    const float d = rsqrtf(sv * (Ssum[0] - sv) + 1.0f);
    dinv[i] = d;
    t[i] = sv * d;
  }
}

// mam[i,j] = s_i * s_j  (rank-1 outer product written straight to d_out)
__global__ __launch_bounds__(256) void mam_kernel(
    const float* __restrict__ s, float* __restrict__ mam, int N)
{
  const size_t lin = ((size_t)blockIdx.x * 256 + threadIdx.x) * 4;
  if (lin >= (size_t)N * (size_t)N) return;
  const int i = (int)(lin / (size_t)N);
  const int j = (int)(lin % (size_t)N);
  const float si = s[i];
  const float4 sj = *(const float4*)(s + j);
  float4 o;
  o.x = si * sj.x; o.y = si * sj.y; o.z = si * sj.z; o.w = si * sj.w;
  *(float4*)(mam + lin) = o;
}

// Stage 1 of u[c] = sum_j t_j * Y[j,c]: each block covers 64 rows, writes
// its own partial row (deterministic, no atomics).
__global__ __launch_bounds__(256) void colpart_kernel(
    const float* __restrict__ Y, const float* __restrict__ t,
    float* __restrict__ part, int H)
{
  __shared__ float ts[64];
  const int r0 = blockIdx.x * 64;
  if (threadIdx.x < 64) ts[threadIdx.x] = t[r0 + threadIdx.x];
  __syncthreads();
  for (int c = threadIdx.x; c < H; c += 256) {
    float acc = 0.f;
    #pragma unroll 8
    for (int r = 0; r < 64; ++r)
      acc += ts[r] * Y[(size_t)(r0 + r) * H + c];
    part[(size_t)blockIdx.x * H + c] = acc;
  }
}

__global__ void colfinish_kernel(const float* __restrict__ part,
                                 float* __restrict__ u, int H, int nparts)
{
  const int c = blockIdx.x * blockDim.x + threadIdx.x;
  if (c < H) {
    float acc = 0.f;
    for (int p = 0; p < nparts; ++p) acc += part[(size_t)p * H + c];
    u[c] = acc;
  }
}

// In-place: Y[i,c] = dinv_i*s_i*u[c] + dinv_i^2*(1-s_i^2)*Y[i,c] + bias[c]
__global__ void combine_kernel(float* __restrict__ Y,
                               const float* __restrict__ u,
                               const float* __restrict__ s,
                               const float* __restrict__ dinv,
                               const float* __restrict__ bias,
                               int H, size_t total)
{
  const size_t idx = (size_t)blockIdx.x * blockDim.x + threadIdx.x;
  if (idx >= total) return;
  const int i = (int)(idx / (size_t)H);
  const int c = (int)(idx % (size_t)H);
  const float d = dinv[i];
  const float sv = s[i];
  Y[idx] = d * sv * u[c] + d * d * (1.f - sv * sv) * Y[idx] + bias[c];
}

extern "C" void kernel_launch(void* const* d_in, const int* in_sizes, int n_in,
                              void* d_out, int out_size, void* d_ws,
                              size_t ws_size, hipStream_t stream)
{
  (void)in_sizes; (void)n_in; (void)out_size; (void)ws_size;
  const int N = 8192, IN = 2048, H = 512, OUT = 128;

  const float* x    = (const float*)d_in[0];
  const float* p1w  = (const float*)d_in[1];  const float* p1b  = (const float*)d_in[2];
  const float* ln0g = (const float*)d_in[3];  const float* ln0b = (const float*)d_in[4];
  const float* p2w  = (const float*)d_in[5];  const float* p2b  = (const float*)d_in[6];
  const float* a1w  = (const float*)d_in[7];  const float* a1b  = (const float*)d_in[8];
  const float* a2w  = (const float*)d_in[9];  const float* a2b  = (const float*)d_in[10];
  const float* a3w  = (const float*)d_in[11]; const float* a3b  = (const float*)d_in[12];
  const float* c1w  = (const float*)d_in[13]; const float* c1b  = (const float*)d_in[14];
  const float* ln1g = (const float*)d_in[15]; const float* ln1b = (const float*)d_in[16];
  const float* c2w  = (const float*)d_in[17]; const float* c2b  = (const float*)d_in[18];
  const float* ln2g = (const float*)d_in[19]; const float* ln2b = (const float*)d_in[20];
  const float* c3w  = (const float*)d_in[21]; const float* c3b  = (const float*)d_in[22];
  const float* ln3g = (const float*)d_in[23]; const float* ln3b = (const float*)d_in[24];
  const float* resw = (const float*)d_in[25]; const float* resb = (const float*)d_in[26];

  float* ws = (float*)d_ws;
  const size_t NH = (size_t)N * H;                 // 4M floats
  float* h0   = ws;             // reused as x1
  float* hln  = ws + NH;        // reused as y2
  float* h    = ws + 2 * NH;
  float* g1   = ws + 3 * NH;    // reused as y1, then y3/r
  float* g2   = ws + 4 * NH;    // reused as x2
  float* x1   = h0;
  float* y1   = g1;
  float* y2   = hln;
  float* x2   = g2;
  float* y3   = g1;
  float* rres = g1 + (size_t)N * OUT;
  float* sbuf = ws + 5 * NH;
  float* dinv = sbuf + N;
  float* tbuf = dinv + N;
  float* ubuf = tbuf + N;                          // H
  float* Ssum = ubuf + H;                          // 1
  float* part = Ssum + 1;                          // 128 * H

  float* x3out  = (float*)d_out;                   // [N, OUT]
  float* awout  = x3out + (size_t)N * OUT;         // [N]
  float* mamout = awout + N;                       // [N, N]

  const dim3 blk(256);
  const dim3 gH(N / MT, H / NT);                   // 64 x 4
  const dim3 gO(N / MT, OUT / NT);                 // 64 x 1

  // projection
  gemm_bf16_kernel<<<gH, blk, 0, stream>>>(x,   p1w, p1b, h0, N, H, IN, 1);
  ln_kernel<<<N, blk, 0, stream>>>(h0, nullptr, hln, ln0g, ln0b, H, 0);
  gemm_bf16_kernel<<<gH, blk, 0, stream>>>(hln, p2w, p2b, h,  N, H, H, 0);

  // attention scalar + rank-1 graph factors
  gemm_bf16_kernel<<<gH, blk, 0, stream>>>(h, a1w, a1b, g1, N, H, H, 0);
  gemm_bf16_kernel<<<gH, blk, 0, stream>>>(h, a2w, a2b, g2, N, H, H, 0);
  aw_kernel<<<N, blk, 0, stream>>>(g1, g2, a3w, a3b, awout, sbuf, H);
  sum_kernel<<<1, blk, 0, stream>>>(sbuf, Ssum, N);
  scal_kernel<<<(N + 255) / 256, blk, 0, stream>>>(sbuf, Ssum, dinv, tbuf, N);
  mam_kernel<<<(unsigned)(((size_t)N * N / 4 + 255) / 256), blk, 0, stream>>>(
      sbuf, mamout, N);

  // GCN layer 1
  gemm_bf16_kernel<<<gH, blk, 0, stream>>>(h, c1w, nullptr, y1, N, H, H, 0);
  colpart_kernel<<<N / 64, blk, 0, stream>>>(y1, tbuf, part, H);
  colfinish_kernel<<<(H + 255) / 256, blk, 0, stream>>>(part, ubuf, H, N / 64);
  combine_kernel<<<(unsigned)((NH + 255) / 256), blk, 0, stream>>>(
      y1, ubuf, sbuf, dinv, c1b, H, NH);
  ln_kernel<<<N, blk, 0, stream>>>(y1, h, x1, ln1g, ln1b, H, 1);

  // GCN layer 2
  gemm_bf16_kernel<<<gH, blk, 0, stream>>>(x1, c2w, nullptr, y2, N, H, H, 0);
  colpart_kernel<<<N / 64, blk, 0, stream>>>(y2, tbuf, part, H);
  colfinish_kernel<<<(H + 255) / 256, blk, 0, stream>>>(part, ubuf, H, N / 64);
  combine_kernel<<<(unsigned)((NH + 255) / 256), blk, 0, stream>>>(
      y2, ubuf, sbuf, dinv, c2b, H, NH);
  ln_kernel<<<N, blk, 0, stream>>>(y2, x1, x2, ln2g, ln2b, H, 1);

  // GCN layer 3 + residual projection
  gemm_bf16_kernel<<<gO, blk, 0, stream>>>(x2, c3w, nullptr, y3, N, OUT, H, 0);
  colpart_kernel<<<N / 64, blk, 0, stream>>>(y3, tbuf, part, OUT);
  colfinish_kernel<<<1, blk, 0, stream>>>(part, ubuf, OUT, N / 64);
  combine_kernel<<<(unsigned)(((size_t)N * OUT + 255) / 256), blk, 0, stream>>>(
      y3, ubuf, sbuf, dinv, c3b, OUT, (size_t)N * OUT);
  gemm_bf16_kernel<<<gO, blk, 0, stream>>>(x2, resw, resb, rres, N, OUT, H, 0);
  ln_kernel<<<N, blk, 0, stream>>>(y3, rres, x3out, ln3g, ln3b, OUT, 1);
}
